// Com_Net_75419625718167
// MI455X (gfx1250) — compile-verified, matrix operand order checked by reading, not run on hardware
//
#include <hip/hip_runtime.h>
#include <hip/hip_bf16.h>
#include <cstddef>

// ---------------------------------------------------------------------------
// CDNA5 (gfx1250) 3D U-Net.
//   * Activations voxel-major, channel-PADDED: [M][Cp], Cp = max(C,32)
//     (always a power of two). fp32 master + bf16 mirror; the mirror is
//     written once by the producer, so the GEMM hot loop has ZERO
//     format conversions.
//   * Weights pre-swizzled to bf16 WMMA B-fragment order (zero-padded K/N).
//   * Implicit-GEMM conv on v_wmma_f32_16x16x32_bf16; tap loop unrolled
//     (template KSIZE). Out-of-bounds taps read a ZERO PAGE instead of
//     being predicated -> the chunk loop is branch-free: 4x b128 + wmma.
//   * BN(training): two-stage per-channel reduction + fused norm/ReLU/add;
//     fp32 epilogues (stats, residual, cls) keep full precision.
// ---------------------------------------------------------------------------

typedef __attribute__((ext_vector_type(16))) __bf16 v16bf;
typedef __attribute__((ext_vector_type(8)))  __bf16 v8bf;
typedef __attribute__((ext_vector_type(8)))  float  v8f;

#define BN_EPS 1e-5f
#define BN_NS  32

// ---------------------------------------------------------------------------
// Weight repack: W fp32 (DHWIO, [t][ci][co]) -> packed bf16 B fragments.
// Packed layout: ((t*nck + ck)*ntile + nt)*512 + lane*16 + j
// K-mapping (16-bit B 32x16): ci = ck*32 + ((j<8)? j : j+8) + khalf*8.
// ci >= Cin / n >= Cout entries are zero-padded.
// ---------------------------------------------------------------------------
__global__ __launch_bounds__(256) void repack_w_kernel(
    const float* __restrict__ W, __bf16* __restrict__ P,
    int Cin, int Cout, int nck, int ntile, int total)
{
  const int idx = blockIdx.x * 256 + threadIdx.x;
  if (idx >= total) return;
  const int j     = idx & 15;
  const int lane  = (idx >> 4) & 31;
  int tmp         = idx >> 9;
  const int nt    = tmp % ntile;  tmp /= ntile;
  const int ck    = tmp % nck;    tmp /= nck;
  const int t     = tmp;
  const int khalf = lane >> 4;
  const int nloc  = lane & 15;
  const int ci    = (ck << 5) + ((j < 8) ? j : j + 8) + khalf * 8;
  const int n     = nt * 16 + nloc;
  float v = 0.f;
  if (ci < Cin && n < Cout) v = W[((size_t)t * Cin + ci) * Cout + n];
  P[idx] = (__bf16)v;
}

// Zero-fill (zero page + any small init), deterministic per launch.
__global__ __launch_bounds__(256) void zero_kernel(float* __restrict__ p, int n)
{
  const int i = blockIdx.x * 256 + threadIdx.x;
  if (i < n) p[i] = 0.f;
}

// ---------------------------------------------------------------------------
// Implicit-GEMM conv: X bf16 [Mi][Cpin] -> Y fp32 [Mo][Cpout].
// One wave owns a 16(M) x 16(N) tile; block = 4 waves = 64 output voxels.
// Out-of-bounds lanes redirect their A-row pointer to the zero page `zb`,
// so the inner loop has no predication and no zero-fill movs.
// up_tap >= 0 (KSIZE==1 instantiation): transpose-conv k2s2 tap; the store
// scatters to the fine grid at offset (a,b,c) = bits of up_tap.
// ---------------------------------------------------------------------------
template <int KSIZE>
__global__ __launch_bounds__(128) void conv3d_wmma(
    const __bf16* __restrict__ X, const __bf16* __restrict__ P,
    const __bf16* __restrict__ zb, float* __restrict__ Y,
    int Cpin, int Cpout,
    int Din, int Hin, int Win,
    int Do_, int Ho, int Wo,
    int stride, int pad, int up_tap)
{
  constexpr int TAPS = KSIZE * KSIZE * KSIZE;
  const int lane   = threadIdx.x & 31;
  const int wid    = threadIdx.x >> 5;
  const int M      = Do_ * Ho * Wo;
  const int mbase  = blockIdx.x * 64 + wid * 16;
  const int ntid   = blockIdx.y;            // N-tile index
  const int ntile  = gridDim.y;
  const int nloc   = lane & 15;
  const int n      = ntid * 16 + nloc;
  const int row    = mbase + nloc;          // this lane's A row
  const int khalf  = lane >> 4;
  const int nck    = Cpin >> 5;             // Cpin is a multiple of 32

  const bool rowok = row < M;
  int zo = 0, yo = 0, xo = 0;
  if (rowok) {
    zo = row / (Ho * Wo);
    const int r2 = row - zo * Ho * Wo;
    yo = r2 / Wo;
    xo = r2 - yo * Wo;
  }

  v8f acc = {0.f, 0.f, 0.f, 0.f, 0.f, 0.f, 0.f, 0.f};

#pragma unroll
  for (int t = 0; t < TAPS; ++t) {
    // ---- per-tap (not per-element) spatial decode; OOB -> zero page ----
    const __bf16* Xb = zb;
    if (rowok) {
      if (up_tap >= 0) {
        Xb = X + (size_t)row * Cpin;
      } else {
        const int kd = t / (KSIZE * KSIZE);
        const int rr = t - kd * KSIZE * KSIZE;
        const int kh = rr / KSIZE;
        const int kw = rr - kh * KSIZE;
        const int zi = zo * stride - pad + kd;
        const int yi = yo * stride - pad + kh;
        const int xi = xo * stride - pad + kw;
        if ((unsigned)zi < (unsigned)Din && (unsigned)yi < (unsigned)Hin &&
            (unsigned)xi < (unsigned)Win)
          Xb = X + ((size_t)(zi * Hin + yi) * Win + xi) * Cpin;
      }
    }
    // prefetch next tap's packed-weight slab (global_prefetch_b8)
    if (t + 1 < TAPS)
      __builtin_prefetch(P + ((((size_t)(t + 1) * nck) * ntile + ntid) << 9), 0, 1);

    for (int ck = 0; ck < nck; ++ck) {
      // ---- B fragment: pre-swizzled bf16, one 32B vector load ----
      const __bf16* Bb =
          P + ((((size_t)t * nck + ck) * ntile + ntid) << 9) + (lane << 4);
      const v16bf b = *(const v16bf*)Bb;

      // ---- A fragment: two contiguous 16B bf16 runs, unconditional ----
      const int c0 = (ck << 5) + khalf * 8;   // elements 0..7  -> ci = c0+j
      const int c1 = c0 + 16;                 // elements 8..15 -> ci = c1+(j-8)
      const v8bf alo = *(const v8bf*)(Xb + c0);
      const v8bf ahi = *(const v8bf*)(Xb + c1);
      const v16bf a = __builtin_shufflevector(
          alo, ahi, 0, 1, 2, 3, 4, 5, 6, 7, 8, 9, 10, 11, 12, 13, 14, 15);

      acc = __builtin_amdgcn_wmma_f32_16x16x32_bf16(
          false, a, false, b, (short)0, acc, false, false);
    }
  }

  // ---- store: C/D layout VGPR r -> M = r (+8 for upper half-wave) ----
#pragma unroll
  for (int r = 0; r < 8; ++r) {
    const int orow = mbase + r + (khalf ? 8 : 0);
    if (orow < M) {
      const float v = acc[r];
      if (up_tap >= 0) {
        const int ua = up_tap >> 2, ub = (up_tap >> 1) & 1, uc = up_tap & 1;
        const int z = orow / (Ho * Wo);
        const int r2 = orow - z * Ho * Wo;
        const int y = r2 / Wo;
        const int x = r2 - y * Wo;
        const size_t fidx =
            ((size_t)(2 * z + ua) * (2 * Ho) + (2 * y + ub)) * (2 * Wo) +
            (2 * x + uc);
        Y[fidx * Cpout + n] = v;
      } else {
        Y[(size_t)orow * Cpout + n] = v;
      }
    }
  }
}

// ---------------------------------------------------------------------------
// Input transpose: x fp32 [C][M] (NCDHW) -> bf16 [M][Cp], pad zeroed.
// ---------------------------------------------------------------------------
__global__ __launch_bounds__(256) void transpose_in_kernel(
    const float* __restrict__ x, __bf16* __restrict__ xT, int C, int Cp, int M)
{
  const int m = blockIdx.x * 256 + threadIdx.x;
  if (m >= M) return;
  for (int c = 0; c < Cp; ++c)
    xT[(size_t)m * Cp + c] = (__bf16)((c < C) ? x[(size_t)c * M + m] : 0.f);
}

// ---------------------------------------------------------------------------
// BN training stats, two-stage. Stage 1: grid (C, BN_NS) slab partials.
// ---------------------------------------------------------------------------
__global__ __launch_bounds__(256) void bn_partial_kernel(
    const float* __restrict__ X, float* __restrict__ ps,
    float* __restrict__ pq, int Cp, int M)
{
  __shared__ float ss[256], sq[256];
  const int c = blockIdx.x, s = blockIdx.y;
  const int chunk = (M + BN_NS - 1) / BN_NS;
  const int m0 = s * chunk;
  const int m1 = (m0 + chunk < M) ? (m0 + chunk) : M;
  float sm = 0.f, q = 0.f;
  for (int m = m0 + (int)threadIdx.x; m < m1; m += 256) {
    const float v = X[(size_t)m * Cp + c];
    sm += v;
    q += v * v;
  }
  ss[threadIdx.x] = sm;
  sq[threadIdx.x] = q;
  __syncthreads();
  for (int st = 128; st > 0; st >>= 1) {
    if ((int)threadIdx.x < st) {
      ss[threadIdx.x] += ss[threadIdx.x + st];
      sq[threadIdx.x] += sq[threadIdx.x + st];
    }
    __syncthreads();
  }
  if (threadIdx.x == 0) {
    ps[c * BN_NS + s] = ss[0];
    pq[c * BN_NS + s] = sq[0];
  }
}

// Stage 2: one wave per channel, shuffle reduce BN_NS partials.
__global__ __launch_bounds__(32) void bn_final_kernel(
    const float* __restrict__ ps, const float* __restrict__ pq,
    float* __restrict__ mean, float* __restrict__ var, int M)
{
  const int c = blockIdx.x;
  float s = ps[c * BN_NS + threadIdx.x];
  float q = pq[c * BN_NS + threadIdx.x];
  for (int off = 16; off > 0; off >>= 1) {
    s += __shfl_down(s, off, 32);
    q += __shfl_down(q, off, 32);
  }
  if (threadIdx.x == 0) {
    const float m = s / (float)M;
    mean[c] = m;
    var[c] = q / (float)M - m * m;  // biased var == jnp.var
  }
}

// ---------------------------------------------------------------------------
// Fused normalize + ReLU + optional residual add (add AFTER relu, matching
// relu(bn(x)) + skip). Writes the fp32 master AND its bf16 mirror. Pad
// channels are written as zero in both, preserving the zero-pad invariant.
// ---------------------------------------------------------------------------
__global__ __launch_bounds__(256) void bn_apply_kernel(
    const float* __restrict__ X, const float* __restrict__ mean,
    const float* __restrict__ var, const float* __restrict__ g,
    const float* __restrict__ b, const float* __restrict__ addsrc,
    float* __restrict__ Y, __bf16* __restrict__ Ybf,
    int C, int cpmask, int total)
{
  const int idx = blockIdx.x * 256 + threadIdx.x;
  if (idx >= total) return;
  const int c = idx & cpmask;
  if (c >= C) {
    Y[idx] = 0.f;
    Ybf[idx] = (__bf16)0.f;
    return;
  }
  float y = (X[idx] - mean[c]) * rsqrtf(var[c] + BN_EPS) * g[c] + b[c];
  y = fmaxf(y, 0.f);
  if (addsrc) y += addsrc[idx];
  Y[idx] = y;
  Ybf[idx] = (__bf16)y;
}

// ---------------------------------------------------------------------------
// 1x1x1 conv, single output channel, fp32. [M][Cp] -> contiguous dot rows.
// ---------------------------------------------------------------------------
__global__ __launch_bounds__(256) void cls_kernel(
    const float* __restrict__ X, const float* __restrict__ w,
    const float* __restrict__ bias, float* __restrict__ out, int C, int Cp,
    int M)
{
  const int m = blockIdx.x * 256 + threadIdx.x;
  if (m >= M) return;
  const float* xr = X + (size_t)m * Cp;
  float s = bias[0];
  for (int c = 0; c < C; c += 4) {
    const float4 xv = *(const float4*)(xr + c);
    const float4 wv = *(const float4*)(w + c);
    s += xv.x * wv.x + xv.y * wv.y + xv.z * wv.z + xv.w * wv.w;
  }
  out[m] = s;
}

// ---------------------------------------------------------------------------
// Host orchestration. Input flattening (JAX tree, dict keys sorted):
//   params.cls      : 4 x [bias, w]                  -> idx 0..7
//   params.dec_res  : 4 x 2 x [b1,b2,g1,g2,w1,w2]    -> idx 8..55
//   params.down     : 5 x [b1,b2,g1,g2,w1,w2]        -> idx 56..85
//   params.enc_res  : 6 x 2 x [b1,b2,g1,g2,w1,w2]    -> idx 86..157
//   params.enc_s1   : [b, g, w]                      -> idx 158..160
//   params.up       : 4 x [b1,b2,g1,g2,w2,wt]        -> idx 161..184
//   x                                                 -> idx 185
// ---------------------------------------------------------------------------
static const int kENC[6] = {16, 32, 64, 128, 256, 512};
static const int kGRD[6] = {64, 32, 16, 8, 4, 2};
static inline int CPAD(int c) { return c < 32 ? 32 : c; }

extern "C" void kernel_launch(void* const* d_in, const int* in_sizes, int n_in,
                              void* d_out, int out_size, void* d_ws,
                              size_t ws_size, hipStream_t stream)
{
  (void)in_sizes; (void)n_in; (void)out_size; (void)ws_size;

  auto F = [&](int i) -> const float* { return (const float*)d_in[i]; };

  // --- workspace bump allocator (float granularity) ---
  float* ws = (float*)d_ws;
  size_t off = 0;
  auto alloc = [&](size_t nelem) { float* p = ws + off; off += nelem; return p; };

  float* mean = alloc(512);
  float* var  = alloc(512);
  float* psum = alloc(512 * BN_NS);
  float* psq  = alloc(512 * BN_NS);
  float* zpage = alloc(256);                            // 512 bf16 zeros (1KB)
  const size_t SMAX = (size_t)32 * 64 * 64 * 64;        // largest padded act
  float*  t0   = alloc(SMAX);                           // conv outputs (fp32)
  float*  t1   = alloc(SMAX);
  float*  t2   = alloc(SMAX);
  __bf16* t1bf = (__bf16*)alloc(SMAX / 2);              // bf16 mirrors
  __bf16* t2bf = (__bf16*)alloc(SMAX / 2);              // also hosts xT
  __bf16* wpack = (__bf16*)alloc(((size_t)27 * 16 * 32 * 512) / 2);
  __bf16* feat5bf = (__bf16*)alloc(((size_t)512 * 8) / 2 + 64);
  float* feat[5];                                       // fp32 skips (s2..s16)
  for (int i = 1; i < 5; ++i)
    feat[i] = alloc((size_t)CPAD(kENC[i]) * kGRD[i] * kGRD[i] * kGRD[i]);
  const __bf16* zb = (const __bf16*)zpage;

  // zero page must be zero on every call (ws is poisoned, not re-zeroed)
  zero_kernel<<<dim3(1), dim3(256), 0, stream>>>(zpage, 256);

  // --- launch helpers ---
  auto run_conv = [&](const __bf16* X, const float* W, float* Y, int Cin,
                      int Cout, int Di, int Hi, int Wi, int Do, int Ho, int Wo,
                      int k, int st, int pad, int tap) {
    const int taps  = (tap >= 0) ? 1 : k * k * k;
    const int Cpin  = CPAD(Cin);
    const int Cpout = CPAD(Cout);
    const int nck   = Cpin / 32;
    const int ntile = Cout / 16;
    const int ptotal = taps * nck * ntile * 512;
    repack_w_kernel<<<dim3((unsigned)((ptotal + 255) / 256)), dim3(256), 0,
                      stream>>>(W, wpack, Cin, Cout, nck, ntile, ptotal);
    dim3 g((unsigned)((Do * Ho * Wo + 63) / 64), (unsigned)ntile);
    if (tap >= 0 || k == 1)
      conv3d_wmma<1><<<g, dim3(128), 0, stream>>>(X, wpack, zb, Y, Cpin, Cpout,
                                                  Di, Hi, Wi, Do, Ho, Wo, st, pad, tap);
    else if (k == 2)
      conv3d_wmma<2><<<g, dim3(128), 0, stream>>>(X, wpack, zb, Y, Cpin, Cpout,
                                                  Di, Hi, Wi, Do, Ho, Wo, st, pad, -1);
    else
      conv3d_wmma<3><<<g, dim3(128), 0, stream>>>(X, wpack, zb, Y, Cpin, Cpout,
                                                  Di, Hi, Wi, Do, Ho, Wo, st, pad, -1);
  };
  auto run_bn = [&](const float* X, int gi, int bi, const float* add, float* Y,
                    __bf16* Ybf, int C, int M) {
    const int Cp = CPAD(C);
    bn_partial_kernel<<<dim3((unsigned)C, BN_NS), dim3(256), 0, stream>>>(
        X, psum, psq, Cp, M);
    bn_final_kernel<<<dim3((unsigned)C), dim3(32), 0, stream>>>(psum, psq,
                                                                mean, var, M);
    const int total = M * Cp;
    bn_apply_kernel<<<dim3((unsigned)((total + 255) / 256)), dim3(256), 0,
                      stream>>>(X, mean, var, F(gi), F(bi), add, Y, Ybf, C,
                                Cp - 1, total);
  };
  // res_p layout at `base`: +0 b1, +1 b2, +2 g1, +3 g2, +4 w1, +5 w2
  // h lives in (t1, t1bf); t0/t2 are scratch.
  auto res = [&](int base, int C, int D) {
    const int M = D * D * D;
    run_conv(t1bf, F(base + 4), t0, C, C, D, D, D, D, D, D, 3, 1, 1, -1);
    run_bn(t0, base + 2, base + 0, nullptr, t2, t2bf, C, M);
    run_conv(t2bf, F(base + 5), t0, C, C, D, D, D, D, D, D, 3, 1, 1, -1);
    run_bn(t0, base + 3, base + 1, t1, t1, t1bf, C, M);   // relu(bn(.)) + h
  };
  auto down = [&](int base, int Ci, int Co, int Di) {
    const int Do = Di / 2, M = Do * Do * Do;
    run_conv(t1bf, F(base + 4), t0, Ci, Co, Di, Di, Di, Do, Do, Do, 2, 2, 0, -1);
    run_bn(t0, base + 2, base + 0, nullptr, t2, t2bf, Co, M);
    run_conv(t2bf, F(base + 5), t0, Co, Co, Do, Do, Do, Do, Do, Do, 3, 1, 1, -1);
    run_bn(t0, base + 3, base + 1, nullptr, t1, t1bf, Co, M);
  };
  // up_p layout at `base`: +0 b1, +1 b2, +2 g1, +3 g2, +4 w2, +5 wt
  auto up = [&](const __bf16* hinbf, const float* skip, int base, int Ci,
                int Co, int Dc) {
    const int Df = 2 * Dc, Mf = Df * Df * Df;
    const float* wt = F(base + 5);
    for (int t = 0; t < 8; ++t)  // 8 disjoint taps tile the fine grid
      run_conv(hinbf, wt + (size_t)t * Ci * Co, t0, Ci, Co, Dc, Dc, Dc, Dc,
               Dc, Dc, 1, 1, 0, t);
    run_bn(t0, base + 2, base + 0, nullptr, t2, t2bf, Co, Mf);
    run_conv(t2bf, F(base + 4), t0, Co, Co, Df, Df, Df, Df, Df, Df, 3, 1, 1, -1);
    run_bn(t0, base + 3, base + 1, skip, t1, t1bf, Co, Mf);  // + skip
  };

  // ================= encoder =================
  {
    const int C = kENC[0], D = kGRD[0], M = D * D * D;
    transpose_in_kernel<<<dim3((unsigned)((M + 255) / 256)), dim3(256), 0,
                          stream>>>(F(185), t2bf, 20, 32, M);
    run_conv(t2bf, F(160), t0, 20, C, D, D, D, D, D, D, 3, 1, 1, -1);
    run_bn(t0, 159, 158, nullptr, t1, t1bf, C, M);
  }
  res(86 + 0 * 6, kENC[0], kGRD[0]);
  res(86 + 1 * 6, kENC[0], kGRD[0]);
  // feats[0] is never consumed by the decoder -> not materialized.
  for (int i = 0; i < 5; ++i) {
    down(56 + i * 6, kENC[i], kENC[i + 1], kGRD[i]);
    res(86 + ((i + 1) * 2 + 0) * 6, kENC[i + 1], kGRD[i + 1]);
    res(86 + ((i + 1) * 2 + 1) * 6, kENC[i + 1], kGRD[i + 1]);
    const int Dn = kGRD[i + 1];
    if (i < 4) {   // fp32 skip features (consumed by decoder adds)
      hipMemcpyAsync(feat[i + 1], t1,
                     (size_t)CPAD(kENC[i + 1]) * Dn * Dn * Dn * sizeof(float),
                     hipMemcpyDeviceToDevice, stream);
    } else {       // bottleneck: only the bf16 mirror is consumed (conv A)
      hipMemcpyAsync(feat5bf, t1bf,
                     (size_t)CPAD(kENC[5]) * Dn * Dn * Dn * sizeof(__bf16),
                     hipMemcpyDeviceToDevice, stream);
    }
  }

  // ================= decoder + heads =================
  float* out = (float*)d_out;
  size_t out_off = 0;
  const __bf16* curbf = feat5bf;
  for (int j = 0; j < 4; ++j) {
    const int Ci = kENC[5 - j], Co = kENC[4 - j];
    const int Dc = kGRD[5 - j], Df = kGRD[4 - j];
    up(curbf, feat[4 - j], 161 + j * 6, Ci, Co, Dc);
    res(8 + (j * 2 + 0) * 6, Co, Df);
    res(8 + (j * 2 + 1) * 6, Co, Df);
    const int M = Df * Df * Df;
    cls_kernel<<<dim3((unsigned)((M + 255) / 256)), dim3(256), 0, stream>>>(
        t1, F(j * 2 + 1), F(j * 2 + 0), out + out_off, Co, CPAD(Co), M);
    out_off += (size_t)M;
    curbf = t1bf;
  }
}
